// SelfAttention_47064251630063
// MI455X (gfx1250) — matrix operand order repaired and checked by reading, not compiled, not run on hardware
//
#include <hip/hip_runtime.h>

// ---------------------------------------------------------------------------
// CDNA5 (gfx1250) wave32 WMMA implementation of dilated windowed attention.
// Round 2: K-tile staging now uses GLOBAL_LOAD_ASYNC_TO_LDS_B128 (ASYNCcnt).
// ---------------------------------------------------------------------------

typedef __attribute__((ext_vector_type(16))) __bf16 v16bf;
typedef __attribute__((ext_vector_type(8)))  __bf16 v8bf;
typedef __attribute__((ext_vector_type(8)))  float  v8f;

#define DIM    1024
#define S_LEN  8704
#define HEADS  16
#define HD     64
#define SEQD   4608   // dilated sequence length (2 sequences)
#define WROWS  9216   // dilated rows total (2*4608)

// ---------------- bf16 helpers (bit-level, no __bf16 arithmetic) -----------
static __device__ __forceinline__ unsigned short f2bf(float f) {
    unsigned u = __builtin_bit_cast(unsigned, f);
    unsigned r = u + 0x7fffu + ((u >> 16) & 1u);   // round-to-nearest-even
    return (unsigned short)(r >> 16);
}
static __device__ __forceinline__ float bf2f(unsigned short h) {
    unsigned u = ((unsigned)h) << 16;
    return __builtin_bit_cast(float, u);
}
static __device__ __forceinline__ v16bf zero_frag() {
    unsigned short z = 0; __bf16 b; __builtin_memcpy(&b, &z, 2);
    v16bf a;
#pragma unroll
    for (int i = 0; i < 16; i++) a[i] = b;
    return a;
}
static __device__ __forceinline__ v8f zero_acc() {
    v8f z;
#pragma unroll
    for (int i = 0; i < 8; i++) z[i] = 0.0f;
    return z;
}

// A-operand (16x32 bf16): lane m=l&15; elems 0-15 = K (kh*16).. contiguous.
static __device__ __forceinline__ v16bf a_frag(const unsigned short* rowp, int kh) {
    v8bf lo = *(const v8bf*)(rowp + kh * 16);
    v8bf hi = *(const v8bf*)(rowp + kh * 16 + 8);
    v16bf a;
#pragma unroll
    for (int i = 0; i < 8; i++) { a[i] = lo[i]; a[i + 8] = hi[i]; }
    return a;
}
// B-operand (32x16 bf16): lane n=l&15; elems 0-15 = K (kh*16).. contiguous.
static __device__ __forceinline__ v16bf b_frag(const unsigned short* p) {
    v8bf lo = *(const v8bf*)(p);
    v8bf hi = *(const v8bf*)(p + 8);
    v16bf b;
#pragma unroll
    for (int i = 0; i < 8; i++) { b[i] = lo[i]; b[i + 8] = hi[i]; }
    return b;
}
#define WMMA_BF16(A, B, C) \
    __builtin_amdgcn_wmma_f32_16x16x32_bf16(false, (A), false, (B), (short)0, (C), false, false)

// ---------------- CDNA5 async global->LDS copy (ASYNCcnt-tracked) ----------
// VDST = wave-relative LDS byte address (low 32 bits of generic &shared);
// VADDR = 64-bit global address.  (ISA 15.18.3 opcode 98)
static __device__ __forceinline__ void async_ld_b128(unsigned short* lds_ptr,
                                                     const unsigned short* gptr) {
    unsigned lds_off = (unsigned)(unsigned long long)lds_ptr;  // addr[31:0] = LDS offset
    asm volatile("global_load_async_to_lds_b128 %0, %1, off"
                 :: "v"(lds_off), "v"(gptr)
                 : "memory");
}
static __device__ __forceinline__ void wait_async0() {
    asm volatile("s_wait_asynccnt 0x0" ::: "memory");
}

// ---------------- wave32 cross-lane reductions via ds_swizzle --------------
template <int MASK>
static __device__ __forceinline__ float swz(float x) {
    int y = __builtin_amdgcn_ds_swizzle(__builtin_bit_cast(int, x), (MASK << 10) | 0x1f);
    return __builtin_bit_cast(float, y);
}
static __device__ __forceinline__ float rowmax16(float x) {
    x = fmaxf(x, swz<1>(x)); x = fmaxf(x, swz<2>(x));
    x = fmaxf(x, swz<4>(x)); x = fmaxf(x, swz<8>(x));
    return x;
}
static __device__ __forceinline__ float rowsum16(float x) {
    x += swz<1>(x); x += swz<2>(x); x += swz<4>(x); x += swz<8>(x);
    return x;
}
static __device__ __forceinline__ float wavesum32(float x) {
    x += swz<1>(x); x += swz<2>(x); x += swz<4>(x); x += swz<8>(x); x += swz<16>(x);
    return x;
}

// ---------------------------------------------------------------------------
// Kernel 1: fp32 -> bf16 convert (x)
// ---------------------------------------------------------------------------
__global__ __launch_bounds__(256) void cvt_f32_to_bf16(const float* __restrict__ in,
                                                       unsigned short* __restrict__ out, int n) {
    int i = blockIdx.x * 256 + threadIdx.x;
    if (i < n) out[i] = f2bf(in[i]);
}

// Kernel 2: W (K x N fp32) -> Wt (N x K bf16)
__global__ __launch_bounds__(256) void transpose_w_kernel(const float* __restrict__ W,
                                                          unsigned short* __restrict__ Wt) {
    int i = blockIdx.x * 256 + threadIdx.x;  // < 1024*1024
    int k = i >> 10, n = i & 1023;
    Wt[(size_t)n * DIM + k] = f2bf(W[(size_t)k * DIM + n]);
}

// ---------------------------------------------------------------------------
// Kernel 3: C_bf16[M,1024] = A_bf16[M,1024] @ W + bias  (Bt = W transposed)
// wg tile 128x256 (8 waves, 2x4, each 64x64 = 4x4 WMMA tiles)
// ---------------------------------------------------------------------------
__global__ __launch_bounds__(256) void gemm_bias_bf16(const unsigned short* __restrict__ A,
                                                      const unsigned short* __restrict__ Bt,
                                                      const float* __restrict__ bias,
                                                      unsigned short* __restrict__ C) {
    const int K = DIM, N = DIM;
    int lane = threadIdx.x & 31, wave = threadIdx.x >> 5;
    int kh = lane >> 4, cl = lane & 15;
    int m0 = blockIdx.x * 128 + (wave & 1) * 64;
    int n0 = blockIdx.y * 256 + (wave >> 1) * 64;

    v8f acc[4][4];
#pragma unroll
    for (int a = 0; a < 4; a++)
#pragma unroll
        for (int b = 0; b < 4; b++) acc[a][b] = zero_acc();

    for (int k0 = 0; k0 < K; k0 += 32) {
        v16bf af[4], bfr[4];
#pragma unroll
        for (int t = 0; t < 4; t++)
            af[t] = a_frag(A + (size_t)(m0 + t * 16 + cl) * K + k0, kh);
#pragma unroll
        for (int t = 0; t < 4; t++)
            bfr[t] = b_frag(Bt + (size_t)(n0 + t * 16 + cl) * K + k0 + kh * 16);
#pragma unroll
        for (int tr = 0; tr < 4; tr++)
#pragma unroll
            for (int tc = 0; tc < 4; tc++)
                acc[tr][tc] = WMMA_BF16(af[tr], bfr[tc], acc[tr][tc]);
    }
#pragma unroll
    for (int tr = 0; tr < 4; tr++)
#pragma unroll
        for (int tc = 0; tc < 4; tc++) {
            int col = n0 + tc * 16 + cl;
            float bv = bias[col];
#pragma unroll
            for (int r = 0; r < 8; r++) {
                int row = m0 + tr * 16 + r + 8 * kh;
                C[(size_t)row * N + col] = f2bf(acc[tr][tc][r] + bv);
            }
        }
}

// ---------------------------------------------------------------------------
// Kernel 4: in-place RMSNorm + RoPE on one bf16 [8704,1024] tensor
// one block (256 thr) per row; 4 columns (2 rope pairs) per thread
// ---------------------------------------------------------------------------
__global__ __launch_bounds__(256) void norm_rope_kernel(unsigned short* __restrict__ buf,
                                                        const float* __restrict__ gvec,
                                                        const float* __restrict__ cosv,
                                                        const float* __restrict__ sinv) {
    int s = blockIdx.x;
    unsigned short* row = buf + (size_t)s * DIM;
    int c0 = threadIdx.x * 4;
    float v0 = bf2f(row[c0]), v1 = bf2f(row[c0 + 1]);
    float v2 = bf2f(row[c0 + 2]), v3 = bf2f(row[c0 + 3]);
    float ss = v0 * v0 + v1 * v1 + v2 * v2 + v3 * v3;
    ss = wavesum32(ss);
    __shared__ float red[8];
    int lane = threadIdx.x & 31, wave = threadIdx.x >> 5;
    if (lane == 0) red[wave] = ss;
    __syncthreads();
    float tot = 0.0f;
#pragma unroll
    for (int i = 0; i < 8; i++) tot += red[i];
    float inv = rsqrtf(tot * (1.0f / 1024.0f) + 1e-6f);
    float a0 = v0 * inv * gvec[c0],     a1 = v1 * inv * gvec[c0 + 1];
    float a2 = v2 * inv * gvec[c0 + 2], a3 = v3 * inv * gvec[c0 + 3];
    int j0 = (c0 & 63) >> 1;  // rope pair index within head (0..31)
    float cA = cosv[(size_t)s * 32 + j0],     sA = sinv[(size_t)s * 32 + j0];
    float cB = cosv[(size_t)s * 32 + j0 + 1], sB = sinv[(size_t)s * 32 + j0 + 1];
    row[c0]     = f2bf(a0 * cA - a1 * sA);
    row[c0 + 1] = f2bf(a0 * sA + a1 * cA);
    row[c0 + 2] = f2bf(a2 * cB - a3 * sB);
    row[c0 + 3] = f2bf(a2 * sB + a3 * cB);
}

// ---------------------------------------------------------------------------
// Kernel 5: dilated windowed flash attention
// grid (9 qblocks, 16 heads, 2 dilated seqs), 512 threads = 16 waves x 32 rows
// Q/K/V read from the (un-dilated) lin buffers via the dilation index map;
// rows mapping past S are zeros (matches reference zero-padding).
// K tile staged via async global->LDS copies; V transposed through VGPRs.
// ---------------------------------------------------------------------------
__global__ __launch_bounds__(512) void attn_kernel(const unsigned short* __restrict__ Q,
                                                   const unsigned short* __restrict__ Kd,
                                                   const unsigned short* __restrict__ V,
                                                   unsigned short* __restrict__ O) {
    int ib = blockIdx.x, h = blockIdx.y, d = blockIdx.z;
    int tid = threadIdx.x;
    int lane = tid & 31, wave = tid >> 5;
    int kh = lane >> 4, cl = lane & 15;

    __shared__ alignas(16) unsigned short Kc[32][64];      // K chunk rows
    __shared__ alignas(16) unsigned short Vt[64][32];      // V chunk transposed
    __shared__ alignas(16) unsigned short Pb[16][32][32];  // per-wave P bounce

    int qbase = ib * 512 + wave * 32;  // dilated query position of this wave

    // Q fragments (scaled later on scores)
    v16bf qf[2][2];
#pragma unroll
    for (int tr = 0; tr < 2; tr++) {
        int p = qbase + tr * 16 + cl;
        int s = (p >> 8) * 512 + d * 256 + (p & 255);
        if (s < S_LEN) {
#pragma unroll
            for (int ks = 0; ks < 2; ks++)
                qf[tr][ks] = a_frag(Q + (size_t)s * DIM + h * HD + ks * 32, kh);
        } else {
            qf[tr][0] = zero_frag();
            qf[tr][1] = zero_frag();
        }
    }

    v8f acc[2][4];
    float mrow[2][8], lrow[2][8];
#pragma unroll
    for (int tr = 0; tr < 2; tr++) {
#pragma unroll
        for (int tv = 0; tv < 4; tv++) acc[tr][tv] = zero_acc();
#pragma unroll
        for (int r = 0; r < 8; r++) { mrow[tr][r] = -1e30f; lrow[tr][r] = 0.0f; }
    }

    int kw0 = (ib - 1) * 512;  // window start (may be negative)
    for (int ch = 0; ch < 48; ch++) {
        int k0 = kw0 + ch * 32;
        if (k0 + 31 < 0 || k0 >= SEQD) continue;  // uniform across block

        // ---- cooperative staging: K via async global->LDS, V via VGPR transpose
        if (tid < 256) {                            // wave-uniform predicate (waves 0-7)
            int r = tid >> 3, cb = (tid & 7) * 8;
            int p = k0 + r;
            int s = (p >= 0 && p < SEQD) ? ((p >> 8) * 512 + d * 256 + (p & 255)) : S_LEN;
            if (s < S_LEN) {
                // async 16B copy straight into LDS (no VGPR roundtrip)
                async_ld_b128(&Kc[r][cb], Kd + (size_t)s * DIM + h * HD + cb);
                v8bf vv = *(const v8bf*)(V + (size_t)s * DIM + h * HD + cb);
                unsigned short tmp[8];
                __builtin_memcpy(tmp, &vv, 16);
#pragma unroll
                for (int e = 0; e < 8; e++) Vt[cb + e][r] = tmp[e];
            } else {
#pragma unroll
                for (int e = 0; e < 8; e++) { Kc[r][cb + e] = 0; Vt[cb + e][r] = 0; }
            }
        }
        wait_async0();     // drain this wave's async copies (ASYNCcnt -> 0)
        __syncthreads();   // publish Kc/Vt to all waves

        // ---- scores: S = Q K^T ----
        v16bf kb[2][2];
#pragma unroll
        for (int tc = 0; tc < 2; tc++)
#pragma unroll
            for (int ks = 0; ks < 2; ks++)
                kb[tc][ks] = b_frag(&Kc[tc * 16 + cl][ks * 32 + kh * 16]);

        v8f sc[2][2];
#pragma unroll
        for (int tr = 0; tr < 2; tr++)
#pragma unroll
            for (int tc = 0; tc < 2; tc++) {
                v8f c = zero_acc();
                c = WMMA_BF16(qf[tr][0], kb[tc][0], c);
                c = WMMA_BF16(qf[tr][1], kb[tc][1], c);
                sc[tr][tc] = c;
            }

        // ---- mask + scale + online softmax (per row, 16-lane reductions) ----
#pragma unroll
        for (int tr = 0; tr < 2; tr++) {
#pragma unroll
            for (int tc = 0; tc < 2; tc++) {
                int kcol = k0 + tc * 16 + cl;
#pragma unroll
                for (int r = 0; r < 8; r++) {
                    int qi = qbase + tr * 16 + r + 8 * kh;
                    float sv = sc[tr][tc][r] * 0.125f;  // 1/sqrt(64)
                    bool valid = (kcol >= 0) && (kcol < SEQD) &&
                                 (qi - kcol <= 512) && (kcol - qi <= 512);
                    sc[tr][tc][r] = valid ? sv : -1e9f;
                }
            }
#pragma unroll
            for (int r = 0; r < 8; r++) {
                float cm = fmaxf(sc[tr][0][r], sc[tr][1][r]);
                cm = rowmax16(cm);
                float mo = mrow[tr][r];
                float mn = fmaxf(mo, cm);
                float corr = __expf(mo - mn);
                float p0 = __expf(sc[tr][0][r] - mn);
                float p1 = __expf(sc[tr][1][r] - mn);
                float psum = rowsum16(p0 + p1);
                lrow[tr][r] = lrow[tr][r] * corr + psum;
                mrow[tr][r] = mn;
#pragma unroll
                for (int tv = 0; tv < 4; tv++) acc[tr][tv][r] *= corr;
                sc[tr][0][r] = p0;
                sc[tr][1][r] = p1;
            }
            // C-layout -> LDS (per-wave region)
#pragma unroll
            for (int tc = 0; tc < 2; tc++)
#pragma unroll
                for (int r = 0; r < 8; r++)
                    Pb[wave][tr * 16 + r + 8 * kh][tc * 16 + cl] = f2bf(sc[tr][tc][r]);
        }
        asm volatile("s_wait_dscnt 0x0" ::: "memory");  // wave-local LDS RAW

        // ---- O += P V ----
        v16bf vf[4];
#pragma unroll
        for (int tv = 0; tv < 4; tv++)
            vf[tv] = b_frag(&Vt[tv * 16 + cl][kh * 16]);
#pragma unroll
        for (int tr = 0; tr < 2; tr++) {
            v16bf pf = a_frag(&Pb[wave][tr * 16 + cl][0], kh);
#pragma unroll
            for (int tv = 0; tv < 4; tv++)
                acc[tr][tv] = WMMA_BF16(pf, vf[tv], acc[tr][tv]);
        }
        __syncthreads();  // before next chunk overwrites Kc/Vt
    }

    // ---- epilogue: O[g, h*64 + col] = acc / l ----
#pragma unroll
    for (int tr = 0; tr < 2; tr++)
#pragma unroll
        for (int tv = 0; tv < 4; tv++)
#pragma unroll
            for (int r = 0; r < 8; r++) {
                int p = qbase + tr * 16 + r + 8 * kh;
                int g = d * SEQD + p;
                float o = acc[tr][tv][r] / lrow[tr][r];
                O[(size_t)g * DIM + h * HD + tv * 16 + cl] = f2bf(o);
            }
}

// ---------------------------------------------------------------------------
// Kernel 6: out[s,:] = o_dil[g,:] @ wo + bo  with inverse-dilation row remap
// ---------------------------------------------------------------------------
__global__ __launch_bounds__(256) void gemm_out_proj(const unsigned short* __restrict__ A,
                                                     const unsigned short* __restrict__ Bt,
                                                     const float* __restrict__ bias,
                                                     float* __restrict__ OUT) {
    const int K = DIM, N = DIM;
    int lane = threadIdx.x & 31, wave = threadIdx.x >> 5;
    int kh = lane >> 4, cl = lane & 15;
    int m0 = blockIdx.x * 128 + (wave & 1) * 64;
    int n0 = blockIdx.y * 256 + (wave >> 1) * 64;

    v8f acc[4][4];
#pragma unroll
    for (int a = 0; a < 4; a++)
#pragma unroll
        for (int b = 0; b < 4; b++) acc[a][b] = zero_acc();

    for (int k0 = 0; k0 < K; k0 += 32) {
        v16bf af[4], bfr[4];
#pragma unroll
        for (int t = 0; t < 4; t++)
            af[t] = a_frag(A + (size_t)(m0 + t * 16 + cl) * K + k0, kh);
#pragma unroll
        for (int t = 0; t < 4; t++)
            bfr[t] = b_frag(Bt + (size_t)(n0 + t * 16 + cl) * K + k0 + kh * 16);
#pragma unroll
        for (int tr = 0; tr < 4; tr++)
#pragma unroll
            for (int tc = 0; tc < 4; tc++)
                acc[tr][tc] = WMMA_BF16(af[tr], bfr[tc], acc[tr][tc]);
    }
#pragma unroll
    for (int tr = 0; tr < 4; tr++)
#pragma unroll
        for (int tc = 0; tc < 4; tc++) {
            int col = n0 + tc * 16 + cl;
            float bv = bias[col];
#pragma unroll
            for (int r = 0; r < 8; r++) {
                int g = m0 + tr * 16 + r + 8 * kh;   // dilated row
                int dd = g / SEQD;
                int p = g - dd * SEQD;
                int s = (p >> 8) * 512 + dd * 256 + (p & 255);
                if (s < S_LEN)
                    OUT[(size_t)s * N + col] = acc[tr][tc][r] + bv;
            }
        }
}

// ---------------------------------------------------------------------------
// Host launch
// ---------------------------------------------------------------------------
extern "C" void kernel_launch(void* const* d_in, const int* in_sizes, int n_in,
                              void* d_out, int out_size, void* d_ws, size_t ws_size,
                              hipStream_t stream) {
    (void)in_sizes; (void)n_in; (void)out_size; (void)ws_size;

    const float* x    = (const float*)d_in[0];
    const float* fcos = (const float*)d_in[1];
    const float* fsin = (const float*)d_in[2];
    const float* wq   = (const float*)d_in[3];
    const float* bq   = (const float*)d_in[4];
    const float* wk   = (const float*)d_in[5];
    const float* bk   = (const float*)d_in[6];
    const float* wv   = (const float*)d_in[7];
    const float* bv   = (const float*)d_in[8];
    const float* wo   = (const float*)d_in[9];
    const float* bo   = (const float*)d_in[10];
    const float* gq   = (const float*)d_in[11];
    const float* gk   = (const float*)d_in[12];

    char* ws = (char*)d_ws;
    const size_t SZ_XB  = (size_t)S_LEN * DIM * 2;
    const size_t SZ_W   = (size_t)DIM * DIM * 2;
    const size_t SZ_LIN = (size_t)S_LEN * DIM * 2;
    unsigned short* xb   = (unsigned short*)(ws);
    unsigned short* wqb  = (unsigned short*)(ws + SZ_XB);
    unsigned short* wkb  = (unsigned short*)(ws + SZ_XB + SZ_W);
    unsigned short* wvb  = (unsigned short*)(ws + SZ_XB + 2 * SZ_W);
    unsigned short* wob  = (unsigned short*)(ws + SZ_XB + 3 * SZ_W);
    unsigned short* linq = (unsigned short*)(ws + SZ_XB + 4 * SZ_W);
    unsigned short* link = (unsigned short*)(ws + SZ_XB + 4 * SZ_W + SZ_LIN);
    unsigned short* linv = (unsigned short*)(ws + SZ_XB + 4 * SZ_W + 2 * SZ_LIN);
    unsigned short* obuf = (unsigned short*)(ws + SZ_XB + 4 * SZ_W + 3 * SZ_LIN);
    // total workspace use: ~98.6 MB

    // 1) conversions
    cvt_f32_to_bf16<<<(S_LEN * DIM + 255) / 256, 256, 0, stream>>>(x, xb, S_LEN * DIM);
    transpose_w_kernel<<<(DIM * DIM) / 256, 256, 0, stream>>>(wq, wqb);
    transpose_w_kernel<<<(DIM * DIM) / 256, 256, 0, stream>>>(wk, wkb);
    transpose_w_kernel<<<(DIM * DIM) / 256, 256, 0, stream>>>(wv, wvb);
    transpose_w_kernel<<<(DIM * DIM) / 256, 256, 0, stream>>>(wo, wob);

    // 2) QKV projections (WMMA GEMM + bias)
    dim3 gg(S_LEN / 128, DIM / 256);
    gemm_bias_bf16<<<gg, 256, 0, stream>>>(xb, wqb, bq, linq);
    gemm_bias_bf16<<<gg, 256, 0, stream>>>(xb, wkb, bk, link);
    gemm_bias_bf16<<<gg, 256, 0, stream>>>(xb, wvb, bv, linv);

    // 3) RMSNorm + RoPE in place on q and k
    norm_rope_kernel<<<S_LEN, 256, 0, stream>>>(linq, gq, fcos, fsin);
    norm_rope_kernel<<<S_LEN, 256, 0, stream>>>(link, gk, fcos, fsin);

    // 4) dilated windowed attention (flash-style, WMMA + async LDS staging)
    dim3 ga(9, HEADS, 2);
    attn_kernel<<<ga, 512, 0, stream>>>(linq, link, linv, obuf);

    // 5) output projection with inverse-dilation remap -> fp32 d_out
    dim3 go(WROWS / 128, DIM / 256);
    gemm_out_proj<<<go, 256, 0, stream>>>(obuf, wob, bo, (float*)d_out);
}
